// LSTM_75058848464919
// MI455X (gfx1250) — matrix operand order repaired and checked by reading, not compiled, not run on hardware
//
#include <hip/hip_runtime.h>
#include <math.h>

// ---- problem constants --------------------------------------------------
#define BB   256      // batch
#define SS   512      // sequence length
#define HH   1024     // hidden
#define G4H  4096     // 4*H (gates g|i|f|o)
#define XX   10       // vocab
#define CC   10       // classes
#define NWG  64       // persistent workgroups (each owns 16 hidden units)

// ---- WMMA fragment types ------------------------------------------------
typedef __attribute__((ext_vector_type(16))) __bf16 v16bf;
typedef __attribute__((ext_vector_type(8)))  __bf16 v8bf;
typedef __attribute__((ext_vector_type(8)))  float  v8f;

union Frag { v16bf v; v8bf h[2]; };

// ---- explicit global (addrspace 1) pointers to force global_load_* ------
typedef const __bf16 __attribute__((address_space(1)))* gc_bf16_p;
typedef       __bf16 __attribute__((address_space(1)))* g_bf16_p;
typedef const float  __attribute__((address_space(1)))* gc_f32_p;
typedef const int    __attribute__((address_space(1)))* gc_i32_p;
typedef const v8bf   __attribute__((address_space(1)))* gc_v8bf_p;

__device__ __forceinline__ gc_v8bf_p g8(const __bf16* p) {
    return (gc_v8bf_p)(unsigned long long)p;
}

// ---- prep: WhT[n][k] = Wh_gate[k][n%H]  (bf16, B-matrix friendly) -------
__global__ void prep_wht(const float* __restrict__ whg, const float* __restrict__ whi,
                         const float* __restrict__ whf, const float* __restrict__ who,
                         __bf16* __restrict__ WhT) {
    long tid = (long)blockIdx.x * blockDim.x + threadIdx.x;   // 0 .. 4M-1
    int n = (int)(tid >> 10);          // WhT row  (0..4095)
    int k = (int)(tid & 1023);         // WhT col  (0..1023)
    int gate = n >> 10;
    int col  = n & 1023;
    const float* src = (gate == 0) ? whg : (gate == 1) ? whi : (gate == 2) ? whf : who;
    WhT[tid] = (__bf16)src[(long)k * HH + col];
}

// ---- prep: Wrow[xi][n] = Wx_gate[xi][n%H] + b_gate[n%H] -----------------
__global__ void prep_wrow(const float* __restrict__ wxg, const float* __restrict__ wxi,
                          const float* __restrict__ wxf, const float* __restrict__ wxo,
                          const float* __restrict__ bg,  const float* __restrict__ bi,
                          const float* __restrict__ bf,  const float* __restrict__ bo,
                          float* __restrict__ Wrow) {
    int tid = blockIdx.x * blockDim.x + threadIdx.x;          // 0 .. 40959
    int xi = tid / G4H;
    int n  = tid % G4H;
    int gate = n >> 10;
    int col  = n & 1023;
    const float* wx = (gate == 0) ? wxg : (gate == 1) ? wxi : (gate == 2) ? wxf : wxo;
    const float* bb = (gate == 0) ? bg  : (gate == 1) ? bi  : (gate == 2) ? bf  : bo;
    Wrow[tid] = wx[xi * HH + col] + bb[col];
}

__global__ void zero_u32(unsigned* __restrict__ p, int n) {
    int i = blockIdx.x * blockDim.x + threadIdx.x;
    if (i < n) p[i] = 0u;
}

// ---- device-wide split barrier (persistent kernel, all WGs resident) ----
__device__ __forceinline__ void grid_barrier(unsigned* bar) {
    __threadfence();                 // release my h stores to device scope
    __syncthreads();
    if (threadIdx.x == 0) {
        unsigned gen = __hip_atomic_load(&bar[1], __ATOMIC_RELAXED, __HIP_MEMORY_SCOPE_AGENT);
        unsigned prev = __hip_atomic_fetch_add(&bar[0], 1u, __ATOMIC_ACQ_REL, __HIP_MEMORY_SCOPE_AGENT);
        if (prev == NWG - 1) {
            __hip_atomic_store(&bar[0], 0u, __ATOMIC_RELAXED, __HIP_MEMORY_SCOPE_AGENT);
            __hip_atomic_fetch_add(&bar[1], 1u, __ATOMIC_RELEASE, __HIP_MEMORY_SCOPE_AGENT);
        } else {
            while (__hip_atomic_load(&bar[1], __ATOMIC_ACQUIRE, __HIP_MEMORY_SCOPE_AGENT) == gen)
                __builtin_amdgcn_s_sleep(2);
        }
    }
    __syncthreads();
    __threadfence();                 // acquire: drop stale cached h lines
}

// ---- persistent LSTM kernel ---------------------------------------------
// grid = 64 WGs x 256 thr (8 waves). WG b owns hidden units [16b,16b+16).
// Wave w owns batch row-tiles 2w, 2w+1 (rows 32w..32w+31).
// c state lives in VGPRs for the whole sequence.
// The WG's WhT slice (4 gates x 1024 K x 16 cols = 128 KB bf16) is staged
// in LDS once; B operands come from ds_load thereafter.
__global__ void __launch_bounds__(256, 1)
lstm_persist(const int* __restrict__ x,
             const __bf16* __restrict__ WhT,
             const float*  __restrict__ Wrow,
             __bf16* __restrict__ hb0, __bf16* __restrict__ hb1,
             unsigned* __restrict__ bar) {
    extern __shared__ __bf16 Blds[];          // 128 KB: [fs][chunk][lane][8]
    const int lane  = threadIdx.x & 31;
    const int wave  = threadIdx.x >> 5;       // 0..7
    const int hlf   = lane >> 4;              // lane half (0/1)
    const int lm    = lane & 15;
    const int hcol0 = blockIdx.x * 16;        // owned hidden-unit base
    const int rb0   = wave * 32;              // row-tile bases
    const int rb1   = wave * 32 + 16;

    // ---- stage the WG's WhT slice into LDS (once) -----------------------
    // fragment-set fs = g*32 + ks; per lane: 16 bf16 of column (hcol0+lm)
    // at K = ks*32 + 8*hlf + {0..7, 16..23}. Stored as two 16B chunks with 16B
    // lane stride (bank-conflict-free for ds_load_b128).
    for (int fs = wave; fs < 128; fs += 8) {
        int g  = fs >> 5;
        int ks = fs & 31;
        gc_v8bf_p pb = g8(WhT + (long)(g * HH + hcol0 + lm) * HH + ks * 32 + 8 * hlf);
        v8bf c0 = pb[0];
        v8bf c1 = pb[2];
        *(v8bf*)&Blds[(long)((fs * 2 + 0) * 32 + lane) * 8] = c0;
        *(v8bf*)&Blds[(long)((fs * 2 + 1) * 32 + lane) * 8] = c1;
    }
    __syncthreads();

    v8f creg[2];
    #pragma unroll
    for (int r = 0; r < 2; ++r)
        #pragma unroll
        for (int v = 0; v < 8; ++v) creg[r][v] = 0.0f;

    for (int t = 0; t < SS; ++t) {
        const __bf16* hr = (t & 1) ? hb1 : hb0;
        __bf16*       hw = (t & 1) ? hb0 : hb1;

        v8f acc[8];                                // [gate*2 + rowtile]
        #pragma unroll
        for (int i = 0; i < 8; ++i) {
            v8f z = {0.f,0.f,0.f,0.f,0.f,0.f,0.f,0.f};
            acc[i] = z;
        }

        // ---- z[:,gates] = h @ Wh : K = 1024 in chunks of 32 -------------
        #pragma unroll 2
        for (int k0 = 0; k0 < HH; k0 += 32) {
            // A fragments: 16x32 bf16 tiles of h (ISA 16-bit A layout)
            Frag a0, a1;
            {
                gc_v8bf_p pa = g8(hr + (rb0 + lm) * HH + k0 + 8 * hlf);
                a0.h[0] = pa[0];
                a0.h[1] = pa[2];    // +16 elements
            }
            {
                gc_v8bf_p pa = g8(hr + (rb1 + lm) * HH + k0 + 8 * hlf);
                a1.h[0] = pa[0];
                a1.h[1] = pa[2];
            }
            const int ks = k0 >> 5;
            #pragma unroll
            for (int g = 0; g < 4; ++g) {
                const int fs = g * 32 + ks;
                Frag bfr;
                bfr.h[0] = *(const v8bf*)&Blds[(long)((fs * 2 + 0) * 32 + lane) * 8];
                bfr.h[1] = *(const v8bf*)&Blds[(long)((fs * 2 + 1) * 32 + lane) * 8];
                acc[2*g]   = __builtin_amdgcn_wmma_f32_16x16x32_bf16(
                                 false, a0.v, false, bfr.v, (short)0, acc[2*g],   false, false);
                acc[2*g+1] = __builtin_amdgcn_wmma_f32_16x16x32_bf16(
                                 false, a1.v, false, bfr.v, (short)0, acc[2*g+1], false, false);
            }
        }

        // ---- epilogue: + (Wx row + b), activations, c/h update ----------
        // C/D layout: lane 0-15: M=v,   N=lane;  lane 16-31: M=v+8, N=lane-16
        gc_i32_p xg   = (gc_i32_p)(unsigned long long)x;
        gc_f32_p wrg  = (gc_f32_p)(unsigned long long)Wrow;
        g_bf16_p hwg  = (g_bf16_p)(unsigned long long)hw;
        #pragma unroll
        for (int r = 0; r < 2; ++r) {
            const int rb = (r == 0) ? rb0 : rb1;
            #pragma unroll
            for (int v = 0; v < 8; ++v) {
                int m   = v + 8 * hlf;
                int row = rb + m;
                int xi  = xg[row * SS + t];
                gc_f32_p wr = wrg + (long)xi * G4H + hcol0 + lm;
                float zg = acc[0 + r][v] + wr[0];
                float zi = acc[2 + r][v] + wr[HH];
                float zf = acc[4 + r][v] + wr[2 * HH];
                float zo = acc[6 + r][v] + wr[3 * HH];
                float gv = tanhf(zg);
                float iv = 1.0f / (1.0f + __expf(-zi));
                float fv = 1.0f / (1.0f + __expf(-zf));
                float ov = 1.0f / (1.0f + __expf(-zo));
                float cn = gv * iv + creg[r][v] * fv;
                creg[r][v] = cn;
                float hn = tanhf(cn) * ov;
                hwg[row * HH + hcol0 + lm] = (__bf16)hn;
            }
        }

        grid_barrier(bar);
    }
}

// ---- final projection + log_softmax (tiny: 256x1024 @ 1024x10) ----------
__global__ void final_proj(const __bf16* __restrict__ h, const float* __restrict__ Wp,
                           const float* __restrict__ bp, float* __restrict__ out) {
    const int b   = blockIdx.x;
    const int tid = threadIdx.x;
    __shared__ float sm[CC * 256];
    float acc[CC];
    #pragma unroll
    for (int c = 0; c < CC; ++c) acc[c] = 0.0f;
    for (int k = tid; k < HH; k += 256) {
        float hv = (float)h[b * HH + k];
        #pragma unroll
        for (int c = 0; c < CC; ++c) acc[c] += hv * Wp[k * CC + c];
    }
    #pragma unroll
    for (int c = 0; c < CC; ++c) sm[c * 256 + tid] = acc[c];
    __syncthreads();
    for (int s = 128; s > 0; s >>= 1) {
        if (tid < s)
            #pragma unroll
            for (int c = 0; c < CC; ++c) sm[c * 256 + tid] += sm[c * 256 + tid + s];
        __syncthreads();
    }
    if (tid == 0) {
        float l[CC], mx = -1e30f;
        #pragma unroll
        for (int c = 0; c < CC; ++c) { l[c] = sm[c * 256] + bp[c]; mx = fmaxf(mx, l[c]); }
        float se = 0.0f;
        #pragma unroll
        for (int c = 0; c < CC; ++c) se += __expf(l[c] - mx);
        float lse = mx + __logf(se);
        #pragma unroll
        for (int c = 0; c < CC; ++c) out[b * CC + c] = l[c] - lse;
    }
}

// ---- host launcher -------------------------------------------------------
extern "C" void kernel_launch(void* const* d_in, const int* in_sizes, int n_in,
                              void* d_out, int out_size, void* d_ws, size_t ws_size,
                              hipStream_t stream) {
    const int*   x    = (const int*)  d_in[0];
    const float* Wx_g = (const float*)d_in[1];
    const float* Wh_g = (const float*)d_in[2];
    const float* b_g  = (const float*)d_in[3];
    const float* Wx_i = (const float*)d_in[4];
    const float* Wh_i = (const float*)d_in[5];
    const float* b_i  = (const float*)d_in[6];
    const float* Wx_f = (const float*)d_in[7];
    const float* Wh_f = (const float*)d_in[8];
    const float* b_f  = (const float*)d_in[9];
    const float* Wx_o = (const float*)d_in[10];
    const float* Wh_o = (const float*)d_in[11];
    const float* b_o  = (const float*)d_in[12];
    const float* Wp   = (const float*)d_in[13];
    const float* bp   = (const float*)d_in[14];
    float* out = (float*)d_out;

    // workspace layout (all offsets 256B-aligned)
    char* ws = (char*)d_ws;
    __bf16*   WhT  = (__bf16*)ws;                                  // 4096*1024*2 = 8 MB
    float*    Wrow = (float*)(ws + (8u << 20));                    // 10*4096*4   = 160 KB
    __bf16*   h0   = (__bf16*)(ws + (8u << 20) + (256u << 10));    // 256*1024*2  = 512 KB
    __bf16*   h1   = h0 + BB * HH;                                 // 512 KB
    unsigned* bar  = (unsigned*)(h1 + BB * HH);                    // 2 u32

    prep_wht <<<(G4H * HH) / 256, 256, 0, stream>>>(Wh_g, Wh_i, Wh_f, Wh_o, WhT);
    prep_wrow<<<(XX * G4H) / 256, 256, 0, stream>>>(Wx_g, Wx_i, Wx_f, Wx_o,
                                                    b_g, b_i, b_f, b_o, Wrow);
    zero_u32 <<<(BB * HH / 2 + 255) / 256, 256, 0, stream>>>((unsigned*)h0, BB * HH / 2);
    zero_u32 <<<1, 32, 0, stream>>>(bar, 2);

    lstm_persist<<<NWG, 256, 128 * 1024, stream>>>(x, WhT, Wrow, h0, h1, bar);

    // S=512 even: last write (t=511) lands in buffer (512)&1 == 0
    final_proj<<<BB, 256, 0, stream>>>(h0, Wp, bp, out);
}
